// MemCopySplitAttr_76854144794783
// MI455X (gfx1250) — matrix operand order repaired and checked by reading, not compiled, not run on hardware
//
#include <hip/hip_runtime.h>

// ---------------- model constants ----------------
#define B_    16
#define S_    400
#define T_    30
#define M_    10
#define LR_   50
#define LS_   20
#define V_    50000
#define E_    300
#define H_    512
#define A_    64
#define EPAD  320           // E padded to multiple of 32
#define DI_   1964          // E + H + 2H + 2A
#define DIP   1984          // DI padded to multiple of 32
#define GRU_K (EPAD + H_)   // 832
#define DEC_K (DIP + H_)    // 2496

typedef _Float16 h8v  __attribute__((ext_vector_type(8)));
typedef _Float16 v16h __attribute__((ext_vector_type(16)));
typedef float    v8f  __attribute__((ext_vector_type(8)));

union Frag16 { v16h v; h8v h[2]; };

// ---------------- input index map (setup_inputs dict order, params flattened in
// insertion order) ----------------
enum {
  IN_SRC = 0, IN_TRG, IN_SRC2, IN_TRG2, IN_USER, IN_PRODUCT, IN_VSIZE,
  IN_UREV, IN_USUM, IN_PREV, IN_PSUM,
  IP_EMBED,
  IP_REV_WIH_F, IP_REV_WHH_F, IP_REV_BIH_F, IP_REV_BHH_F,
  IP_REV_WIH_B, IP_REV_WHH_B, IP_REV_BIH_B, IP_REV_BHH_B,
  IP_SUM_WIH_F, IP_SUM_WHH_F, IP_SUM_BIH_F, IP_SUM_BHH_F,
  IP_SUM_WIH_B, IP_SUM_WHH_B, IP_SUM_BIH_B, IP_SUM_BHH_B,
  IP_TF_W, IP_TF_B, IP_RS1_W, IP_RS1_B, IP_RS2_W, IP_RS2_B,
  IP_NQ_W, IP_NQ_B, IP_MF_W, IP_MF_B,
  IP_USER_EMBED, IP_PRODUCT_EMBED,
  IP_EA_W, IP_EA_B, IP_EG_W, IP_EG_B,
  IP_DEC_WIH, IP_DEC_WHH, IP_DEC_BIH, IP_DEC_BHH,
  IP_ATT_K, IP_ATT_Q, IP_ATT_E, IP_ATTA_K, IP_ATTA_Q, IP_ATTA_E,
  IP_TC_W, IP_TC_B, IP_AC_W, IP_AC_B, IP_CG_W, IP_CG_B,
  IP_CHD_W, IP_CHD_B, IP_GP_W, IP_GP_B, IP_GEN_W
};

// ---------------- epilogue ----------------
__device__ inline float apply_epi(float v, int e) {
  if (e == 1) return tanhf(v);
  if (e == 2) return 1.f / (1.f + __expf(-v));
  if (e == 3) return v > 0.f ? v : 0.01f * v;   // leaky_relu
  return v;
}

// ---------------- WMMA GEMM: C[M,N] = concat(A0,A1)[M,K] * W[N,K]^T (+bias, epi)
// A0 covers k in [0,K0), A1 covers [K0,K). K0, K multiples of 32; M multiple of 16.
// Block = 128 threads (4 waves); each wave computes a 16x32 output tile pair.
// Branchless inner loops: second tile is clamped to a duplicate when OOB and only
// the store is guarded, so EXEC stays all-1s and no per-iteration exec masking.
__global__ __launch_bounds__(128)
void k_gemm(const _Float16* __restrict__ A0, int K0, int lda0,
            const _Float16* __restrict__ A1, int lda1,
            const _Float16* __restrict__ W, int ldw,
            const float* __restrict__ bias,
            float* __restrict__ C, int ldc,
            int M, int N, int K, int epi)
{
  int wave = threadIdx.x >> 5;
  int lane = threadIdx.x & 31;
  int m0 = blockIdx.x * 16;
  int n0 = (blockIdx.y * 4 + wave) * 32;
  if (n0 >= N) return;
  bool two = (n0 + 16) < N;
  int n1c = two ? (n0 + 16) : n0;   // clamped duplicate when single-tile

  int arow = lane & 15;
  int akb  = (lane >> 4) * 8;    // A per-lane K sub-offsets: akb, akb+16
  int bcol = lane & 15;
  int bkb  = (lane >> 4) * 8;    // B per-lane K sub-offsets: bkb, bkb+16

  const _Float16* ap  = A0 + (size_t)(m0 + arow) * lda0 + akb;
  const _Float16* bp0 = W + (size_t)(n0 + bcol) * ldw + bkb;
  const _Float16* bp1 = W + (size_t)(n1c + bcol) * ldw + bkb;

  v8f c0 = {}; v8f c1 = {};
  int k = 0;
#pragma unroll 2
  for (; k < K0; k += 32) {
    Frag16 a, b0, b1;
    a.h[0]  = *(const h8v*)(ap + k);
    a.h[1]  = *(const h8v*)(ap + k + 16);
    b0.h[0] = *(const h8v*)(bp0 + k);
    b0.h[1] = *(const h8v*)(bp0 + k + 16);
    b1.h[0] = *(const h8v*)(bp1 + k);
    b1.h[1] = *(const h8v*)(bp1 + k + 16);
    c0 = __builtin_amdgcn_wmma_f32_16x16x32_f16(false, a.v, false, b0.v,
                                                (short)0, c0, false, false);
    c1 = __builtin_amdgcn_wmma_f32_16x16x32_f16(false, a.v, false, b1.v,
                                                (short)0, c1, false, false);
  }
  if (k < K) {
    const _Float16* ap1 = A1 + (size_t)(m0 + arow) * lda1 + akb - K0;
#pragma unroll 2
    for (; k < K; k += 32) {
      Frag16 a, b0, b1;
      a.h[0]  = *(const h8v*)(ap1 + k);
      a.h[1]  = *(const h8v*)(ap1 + k + 16);
      b0.h[0] = *(const h8v*)(bp0 + k);
      b0.h[1] = *(const h8v*)(bp0 + k + 16);
      b1.h[0] = *(const h8v*)(bp1 + k);
      b1.h[1] = *(const h8v*)(bp1 + k + 16);
      c0 = __builtin_amdgcn_wmma_f32_16x16x32_f16(false, a.v, false, b0.v,
                                                  (short)0, c0, false, false);
      c1 = __builtin_amdgcn_wmma_f32_16x16x32_f16(false, a.v, false, b1.v,
                                                  (short)0, c1, false, false);
    }
  }
  // C/D layout: lane&15 = N col, vgpr r -> M = r + (lane>>4)*8
  int nC = lane & 15;
  int mbase = (lane >> 4) * 8;
  float bias0 = bias ? bias[n0 + nC] : 0.f;
  float bias1 = bias ? bias[n1c + nC] : 0.f;
#pragma unroll
  for (int r = 0; r < 8; r++) {
    int m = m0 + mbase + r;
    C[(size_t)m * ldc + n0 + nC] = apply_epi(c0[r] + bias0, epi);
    if (two)
      C[(size_t)m * ldc + n0 + 16 + nC] = apply_epi(c1[r] + bias1, epi);
  }
}

// ---------------- GRU gate nonlinearity ----------------
// pre layout per row: [rz_combined(1024) | inn(512) | hn(512)]
__global__ void k_gru_gate(const float* __restrict__ pre,
                           _Float16* __restrict__ h16, float* __restrict__ h32,
                           float* __restrict__ y, int yb_stride, int yoff, int R)
{
  int i = blockIdx.x * blockDim.x + threadIdx.x;
  if (i >= R * H_) return;
  int b = i >> 9, j = i & (H_ - 1);
  const float* p = pre + (size_t)b * 2048;
  float r = 1.f / (1.f + __expf(-p[j]));
  float z = 1.f / (1.f + __expf(-p[512 + j]));
  float n = tanhf(p[1024 + j] + r * p[1536 + j]);
  float hp = h32[i];
  float h = (1.f - z) * n + z * hp;
  h32[i] = h;
  h16[i] = (_Float16)h;
  if (y) y[(size_t)b * yb_stride + yoff + j] = h;
}

// ---------------- build fused GRU weight [2048 x (Ip+512)] + bias[2048] ----------
__global__ void k_build_wcat(const float* __restrict__ Wih, const float* __restrict__ Whh,
                             const float* __restrict__ bih, const float* __restrict__ bhh,
                             _Float16* __restrict__ Wc, float* __restrict__ bc,
                             int I, int Ip)
{
  int ld = Ip + H_;
  size_t total = (size_t)2048 * ld;
  size_t stride = (size_t)gridDim.x * blockDim.x;
  for (size_t idx = (size_t)blockIdx.x * blockDim.x + threadIdx.x; idx < total; idx += stride) {
    int r = (int)(idx / ld), c = (int)(idx % ld);
    float v = 0.f;
    if (r < 1024) {                       // r,z rows: [Wih | Whh]
      if (c < I) v = Wih[(size_t)r * I + c];
      else if (c >= Ip) v = Whh[(size_t)r * H_ + (c - Ip)];
    } else if (r < 1536) {                // inn rows: [Wih_n | 0]
      if (c < I) v = Wih[(size_t)r * I + c];
    } else {                              // hn rows: [0 | Whh_n]
      if (c >= Ip) v = Whh[(size_t)(1024 + (r - 1536)) * H_ + (c - Ip)];
    }
    Wc[idx] = (_Float16)v;
  }
  for (int r = blockIdx.x * blockDim.x + threadIdx.x; r < 2048; r += (int)stride) {
    float v;
    if (r < 1024) v = bih[r] + bhh[r];
    else if (r < 1536) v = bih[r];
    else v = bhh[1024 + (r - 1536)];
    bc[r] = v;
  }
}

// ---------------- f32 -> f16 cast with column pad ----------------
__global__ void k_cast_pad(const float* __restrict__ src, _Float16* __restrict__ dst,
                           int rows, int cols, int colsp)
{
  size_t total = (size_t)rows * colsp;
  size_t i = (size_t)blockIdx.x * blockDim.x + threadIdx.x;
  if (i >= total) return;
  int r = (int)(i / colsp), c = (int)(i % colsp);
  dst[i] = (_Float16)(c < cols ? src[(size_t)r * cols + c] : 0.f);
}

// ---------------- embedding gather -> time-major f16 [Slen][Rtot][EPAD] ----------
__global__ void k_gather_tm(const float* __restrict__ emb, const int* __restrict__ toks,
                            _Float16* __restrict__ X, int rows, int Rtot, int roff, int Slen)
{
  size_t total = (size_t)Slen * rows * EPAD;
  size_t i = (size_t)blockIdx.x * blockDim.x + threadIdx.x;
  if (i >= total) return;
  int s = (int)(i / ((size_t)rows * EPAD));
  int rem = (int)(i % ((size_t)rows * EPAD));
  int b = rem / EPAD, c = rem % EPAD;
  int tok = toks[(size_t)b * Slen + s];
  float v = (c < E_) ? emb[(size_t)tok * E_ + c] : 0.f;
  X[((size_t)s * Rtot + roff + b) * EPAD + c] = (_Float16)v;
}

// ---------------- concat helpers ----------------
__global__ void k_concat2_f16(const float* a, int wa, const float* b, int wb,
                              _Float16* dst, int rows)
{
  int W = wa + wb;
  int i = blockIdx.x * blockDim.x + threadIdx.x;
  if (i >= rows * W) return;
  int r = i / W, c = i % W;
  float v = (c < wa) ? a[(size_t)r * wa + c] : b[(size_t)r * wb + (c - wa)];
  dst[i] = (_Float16)v;
}
__global__ void k_concat2_f32(const float* a, int wa, const float* b, int wb,
                              float* dst, int rows)
{
  int W = wa + wb;
  int i = blockIdx.x * blockDim.x + threadIdx.x;
  if (i >= rows * W) return;
  int r = i / W, c = i % W;
  dst[i] = (c < wa) ? a[(size_t)r * wa + c] : b[(size_t)r * wb + (c - wa)];
}

// ---------------- memory-network attention: ks = softmax_m(rs1(q)+rs2(rev)); mem = ks.sum ----
__global__ void k_mem_attn(const float* __restrict__ q, const float* __restrict__ revf,
                           const float* __restrict__ sumf,
                           const float* rs1W, const float* rs1b,
                           const float* rs2W, const float* rs2b,
                           float* __restrict__ mem)
{
  int b = blockIdx.x, tid = threadIdx.x;
  __shared__ float red[256];
  __shared__ float sc[M_];
  __shared__ float ks[M_];
  float acc = 0.f;
  for (int c = tid; c < 2 * H_; c += 256) acc += rs1W[c] * q[(size_t)b * 2 * H_ + c];
  red[tid] = acc; __syncthreads();
  for (int s = 128; s > 0; s >>= 1) { if (tid < s) red[tid] += red[tid + s]; __syncthreads(); }
  float s1 = red[0] + rs1b[0];
  __syncthreads();
  for (int m = 0; m < M_; m++) {
    float a2 = 0.f;
    const float* rv = revf + ((size_t)(b * M_ + m)) * 2 * H_;
    for (int c = tid; c < 2 * H_; c += 256) a2 += rs2W[c] * rv[c];
    red[tid] = a2; __syncthreads();
    for (int s = 128; s > 0; s >>= 1) { if (tid < s) red[tid] += red[tid + s]; __syncthreads(); }
    if (tid == 0) sc[m] = s1 + red[0] + rs2b[0];
    __syncthreads();
  }
  if (tid == 0) {
    float mx = -1e30f;
    for (int m = 0; m < M_; m++) mx = fmaxf(mx, sc[m]);
    float s = 0.f;
    for (int m = 0; m < M_; m++) { ks[m] = __expf(sc[m] - mx); s += ks[m]; }
    for (int m = 0; m < M_; m++) ks[m] /= s;
  }
  __syncthreads();
  for (int d = tid; d < 2 * H_; d += 256) {
    float a = 0.f;
    for (int m = 0; m < M_; m++) a += ks[m] * sumf[((size_t)(b * M_ + m)) * 2 * H_ + d];
    mem[(size_t)b * 2 * H_ + d] = a;
  }
}

// ---------------- per-batch sigmoid(dot(concat(pieces), w) + bias) ----------------
__global__ void k_dotcat(const float* __restrict__ wv, const float* __restrict__ bias,
                         const float* p0, int w0, const float* p1, int w1,
                         const float* p2, int w2, const float* p3, int w3,
                         float* out)
{
  int b = blockIdx.x, tid = threadIdx.x;
  __shared__ float red[256];
  float acc = 0.f; int off = 0;
  for (int c = tid; c < w0; c += 256) acc += wv[c] * p0[(size_t)b * w0 + c];
  off = w0;
  if (p1) { for (int c = tid; c < w1; c += 256) acc += wv[off + c] * p1[(size_t)b * w1 + c]; off += w1; }
  if (p2) { for (int c = tid; c < w2; c += 256) acc += wv[off + c] * p2[(size_t)b * w2 + c]; off += w2; }
  if (p3) { for (int c = tid; c < w3; c += 256) acc += wv[off + c] * p3[(size_t)b * w3 + c]; off += w3; }
  red[tid] = acc; __syncthreads();
  for (int s = 128; s > 0; s >>= 1) { if (tid < s) red[tid] += red[tid + s]; __syncthreads(); }
  if (tid == 0) out[b] = 1.f / (1.f + __expf(-(red[0] + bias[0])));
}

// ---------------- misc small kernels ----------------
__global__ void k_attrs(const float* ue, const float* pe, const int* user,
                        const int* prod, float* attrs)
{
  int i = blockIdx.x * blockDim.x + threadIdx.x;
  if (i >= B_ * 2 * A_) return;
  int b = i / (2 * A_), c = i % (2 * A_);
  attrs[i] = (c < A_) ? ue[(size_t)user[b] * A_ + c] : pe[(size_t)prod[b] * A_ + (c - A_)];
}

__global__ void k_h0(const float* g, const float* tf, const float* af,
                     float* h32, _Float16* h16)
{
  int i = blockIdx.x * blockDim.x + threadIdx.x;
  if (i >= B_ * H_) return;
  int b = i >> 9;
  float h = g[b] * tf[i] + (1.f - g[b]) * af[i];
  h32[i] = h; h16[i] = (_Float16)h;
}

__global__ void k_upm(const float* attrs, const float* memout, float* upm)
{
  int i = blockIdx.x * blockDim.x + threadIdx.x;
  if (i >= B_ * 1152) return;
  int b = i / 1152, c = i % 1152;
  upm[i] = (c < 128) ? attrs[(size_t)b * 128 + c] : memout[(size_t)b * 1024 + (c - 128)];
}

__global__ void k_dec_build(const float* __restrict__ emb, const int* __restrict__ trg, int t,
                            const float* __restrict__ ch, const float* __restrict__ upm,
                            _Float16* __restrict__ A0, float* __restrict__ prevemb)
{
  int i = blockIdx.x * blockDim.x + threadIdx.x;
  if (i >= B_ * DIP) return;
  int b = i / DIP, c = i % DIP;
  float v;
  if (c < E_) {
    int tok = trg[(size_t)b * T_ + t];
    v = emb[(size_t)tok * E_ + c];
    prevemb[(size_t)b * E_ + c] = v;
  } else if (c < E_ + H_) {
    v = ch[(size_t)b * H_ + (c - E_)];
  } else if (c < DI_) {
    v = upm[(size_t)b * 1152 + (c - E_ - H_)];
  } else v = 0.f;
  A0[i] = (_Float16)v;
}

__global__ void k_attn_scores(const float* __restrict__ qp, const float* __restrict__ pk,
                              const float* __restrict__ att_e, const int* __restrict__ src,
                              float* __restrict__ sc)
{
  int b = blockIdx.x;
  int s = blockIdx.y * 8 + (threadIdx.x >> 5);
  int lane = threadIdx.x & 31;
  const float* pkr = pk + ((size_t)b * S_ + s) * H_;
  const float* qr = qp + (size_t)b * H_;
  float acc = 0.f;
  for (int h = lane; h < H_; h += 32) acc += tanhf(qr[h] + pkr[h]) * att_e[h];
  for (int m = 16; m > 0; m >>= 1) acc += __shfl_xor(acc, m, 32);
  if (lane == 0) {
    if (src[(size_t)b * S_ + s] == 0) acc = -1e30f;
    sc[(size_t)b * S_ + s] = acc;
  }
}

__global__ void k_softmax_row(const float* __restrict__ sc, float* __restrict__ al, int n)
{
  int b = blockIdx.x, tid = threadIdx.x;
  __shared__ float red[256];
  float mx = -1e30f;
  for (int i = tid; i < n; i += 256) mx = fmaxf(mx, sc[(size_t)b * n + i]);
  red[tid] = mx; __syncthreads();
  for (int s = 128; s > 0; s >>= 1) { if (tid < s) red[tid] = fmaxf(red[tid], red[tid + s]); __syncthreads(); }
  mx = red[0]; __syncthreads();
  float sum = 0.f;
  for (int i = tid; i < n; i += 256) sum += __expf(sc[(size_t)b * n + i] - mx);
  red[tid] = sum; __syncthreads();
  for (int s = 128; s > 0; s >>= 1) { if (tid < s) red[tid] += red[tid + s]; __syncthreads(); }
  float inv = 1.f / red[0];
  for (int i = tid; i < n; i += 256)
    al[(size_t)b * n + i] = __expf(sc[(size_t)b * n + i] - mx) * inv;
}

__global__ void k_attn_ctx(const float* __restrict__ al, const float* __restrict__ enc,
                           float* __restrict__ ctx)
{
  int b = blockIdx.x;
  int d = blockIdx.y * 128 + threadIdx.x;
  float a = 0.f;
  for (int s = 0; s < S_; s++)
    a += al[(size_t)b * S_ + s] * enc[((size_t)b * S_ + s) * (2 * H_) + d];
  ctx[(size_t)b * 2 * H_ + d] = a;
}

__global__ void k_attr_attn(const float* __restrict__ qpa, const float* __restrict__ pka,
                            const float* __restrict__ atta_e, const float* __restrict__ attrs,
                            float* __restrict__ ctx_a)
{
  int b = blockIdx.x, tid = threadIdx.x;      // 64 threads, 2 waves
  int k = tid >> 5, lane = tid & 31;
  const float* pr = pka + ((size_t)(b * 2 + k)) * H_;
  float acc = 0.f;
  for (int h = lane; h < H_; h += 32) acc += tanhf(qpa[(size_t)b * H_ + h] + pr[h]) * atta_e[h];
  for (int m = 16; m > 0; m >>= 1) acc += __shfl_xor(acc, m, 32);
  __shared__ float s2[2];
  if (lane == 0) s2[k] = acc;
  __syncthreads();
  float mx = fmaxf(s2[0], s2[1]);
  float e0 = __expf(s2[0] - mx), e1 = __expf(s2[1] - mx);
  float inv = 1.f / (e0 + e1);
  float a0 = e0 * inv, a1 = e1 * inv;
  if (tid < A_)
    ctx_a[(size_t)b * A_ + tid] = a0 * attrs[(size_t)b * 128 + tid]
                                + a1 * attrs[(size_t)b * 128 + A_ + tid];
}

__global__ void k_ctx_mix(const float* cg, const float* tc, const float* ac, float* ctx)
{
  int i = blockIdx.x * blockDim.x + threadIdx.x;
  if (i >= B_ * H_) return;
  int b = i >> 9;
  float g = cg[b];
  ctx[i] = g * tc[i] + (1.f - g) * ac[i];
}

__global__ void k_softmax_mix(const float* __restrict__ logits, const float* __restrict__ gp,
                              float* __restrict__ out, int t)
{
  int b = blockIdx.x, tid = threadIdx.x;
  __shared__ float red[256];
  const float* L = logits + (size_t)b * V_;
  float mx = -1e30f;
  for (int v = tid; v < V_; v += 256) mx = fmaxf(mx, L[v]);
  red[tid] = mx; __syncthreads();
  for (int s = 128; s > 0; s >>= 1) { if (tid < s) red[tid] = fmaxf(red[tid], red[tid + s]); __syncthreads(); }
  mx = red[0]; __syncthreads();
  float sum = 0.f;
  for (int v = tid; v < V_; v += 256) sum += __expf(L[v] - mx);
  red[tid] = sum; __syncthreads();
  for (int s = 128; s > 0; s >>= 1) { if (tid < s) red[tid] += red[tid + s]; __syncthreads(); }
  float inv = gp[b] / red[0];
  float* o = out + (size_t)b * T_ * V_ + (size_t)t * V_;
  for (int v = tid; v < V_; v += 256) o[v] = __expf(L[v] - mx) * inv;
}

__global__ void k_copy_scatter(const float* __restrict__ al, const float* __restrict__ gp,
                               const int* __restrict__ src, float* __restrict__ out, int t)
{
  int b = blockIdx.x;
  float w = 1.f - gp[b];
  float* o = out + (size_t)b * T_ * V_ + (size_t)t * V_;
  for (int s = threadIdx.x; s < S_; s += blockDim.x)
    atomicAdd(&o[src[(size_t)b * S_ + s]], w * al[(size_t)b * S_ + s]);
}

// ---------------- host-side launch helpers ----------------
static inline void gemm_launch(hipStream_t st,
                               const _Float16* A0, int K0, int lda0,
                               const _Float16* A1, int lda1,
                               const _Float16* W, int ldw, const float* bias,
                               float* C, int ldc, int M, int N, int K, int epi)
{
  dim3 g(M / 16, (N + 127) / 128);
  k_gemm<<<g, 128, 0, st>>>(A0, K0, lda0, A1, lda1, W, ldw, bias, C, ldc, M, N, K, epi);
}
static inline int nb(size_t n) { return (int)((n + 255) / 256); }

extern "C" void kernel_launch(void* const* d_in, const int* in_sizes, int n_in,
                              void* d_out, int out_size, void* d_ws, size_t ws_size,
                              hipStream_t stream)
{
  (void)in_sizes; (void)n_in; (void)out_size; (void)ws_size;
  const int* src    = (const int*)d_in[IN_SRC];
  const int* trg_   = (const int*)d_in[IN_TRG2];
  const int* user   = (const int*)d_in[IN_USER];
  const int* prod   = (const int*)d_in[IN_PRODUCT];
  const int* u_rev  = (const int*)d_in[IN_UREV];
  const int* u_sum  = (const int*)d_in[IN_USUM];
  const int* p_rev  = (const int*)d_in[IN_PREV];
  const int* p_sum  = (const int*)d_in[IN_PSUM];
  const float* P_emb = (const float*)d_in[IP_EMBED];
  float* out = (float*)d_out;

  // ---- workspace bump allocator ----
  size_t off = 0;
  auto alloc = [&](size_t bytes) -> void* {
    void* p = (char*)d_ws + off;
    off = (off + bytes + 255) & ~(size_t)255;
    return p;
  };
  auto af32 = [&](size_t n) { return (float*)alloc(n * 4); };
  auto af16 = [&](size_t n) { return (_Float16*)alloc(n * 2); };

  // fused GRU weights
  _Float16* wc_rev_f = af16((size_t)2048 * GRU_K); float* bc_rev_f = af32(2048);
  _Float16* wc_rev_b = af16((size_t)2048 * GRU_K); float* bc_rev_b = af32(2048);
  _Float16* wc_sum_f = af16((size_t)2048 * GRU_K); float* bc_sum_f = af32(2048);
  _Float16* wc_sum_b = af16((size_t)2048 * GRU_K); float* bc_sum_b = af32(2048);
  _Float16* wc_dec   = af16((size_t)2048 * DEC_K); float* bc_dec   = af32(2048);
  // f16 staged dense weights
  _Float16* w_tf   = af16((size_t)H_ * 2 * H_);
  _Float16* w_mf   = af16((size_t)2 * H_ * 4 * H_);
  _Float16* w_ea   = af16((size_t)H_ * 2 * A_);
  _Float16* w_attk = af16((size_t)H_ * 2 * H_);
  _Float16* w_attq = af16((size_t)H_ * H_);
  _Float16* w_attak = af16((size_t)H_ * A_);
  _Float16* w_attaq = af16((size_t)H_ * H_);
  _Float16* w_tc   = af16((size_t)H_ * 2 * H_);
  _Float16* w_ac   = af16((size_t)H_ * A_);
  _Float16* w_chd  = af16((size_t)H_ * 2 * H_);
  _Float16* w_gen  = af16((size_t)V_ * H_);
  // time-major gathered embeddings
  _Float16* X_src = af16((size_t)S_ * B_ * EPAD);
  _Float16* X_rev = af16((size_t)LR_ * 320 * EPAD);
  _Float16* X_sum = af16((size_t)LS_ * 320 * EPAD);
  // recurrence state
  _Float16* h16_mf = af16((size_t)B_ * H_);  float* h32_mf = af32((size_t)B_ * H_);
  _Float16* h16_mb = af16((size_t)B_ * H_);  float* h32_mb = af32((size_t)B_ * H_);
  _Float16* h16_rf = af16((size_t)320 * H_); float* h32_rf = af32((size_t)320 * H_);
  _Float16* h16_rb = af16((size_t)320 * H_); float* h32_rb = af32((size_t)320 * H_);
  _Float16* h16_sf = af16((size_t)320 * H_); float* h32_sf = af32((size_t)320 * H_);
  _Float16* h16_sb = af16((size_t)320 * H_); float* h32_sb = af32((size_t)320 * H_);
  float* preact = af32((size_t)320 * 2048);
  // encoder outputs
  float* enc_hid      = af32((size_t)B_ * S_ * 2 * H_);
  _Float16* enc_hid16 = af16((size_t)B_ * S_ * 2 * H_);
  float* pk           = af32((size_t)B_ * S_ * H_);
  float* enc_final    = af32((size_t)B_ * 2 * H_);
  _Float16* enc_final16 = af16((size_t)B_ * 2 * H_);
  float* rev_final = af32((size_t)320 * 2 * H_);
  float* sum_final = af32((size_t)320 * 2 * H_);
  float* u_mem = af32((size_t)B_ * 2 * H_);
  float* p_mem = af32((size_t)B_ * 2 * H_);
  _Float16* mem_cat16 = af16((size_t)B_ * 4 * H_);
  float* mem_out = af32((size_t)B_ * 2 * H_);
  float* attrs = af32((size_t)B_ * 2 * A_);
  _Float16* attrs16 = af16((size_t)B_ * 2 * A_);
  float* attr_final = af32((size_t)B_ * H_);
  float* text_final = af32((size_t)B_ * H_);
  float* gbuf = af32(B_);
  float* upm = af32((size_t)B_ * 1152);
  float* pka = af32((size_t)2 * B_ * H_);
  // decoder state
  _Float16* h_dec16 = af16((size_t)B_ * H_); float* h_dec32 = af32((size_t)B_ * H_);
  _Float16* decA016 = af16((size_t)B_ * DIP);
  float* prevemb = af32((size_t)B_ * E_);
  float* qp  = af32((size_t)B_ * H_);
  float* qpa = af32((size_t)B_ * H_);
  float* sc = af32((size_t)B_ * S_);
  float* al = af32((size_t)B_ * S_);
  float* ctx_t = af32((size_t)B_ * 2 * H_); _Float16* ctx_t16 = af16((size_t)B_ * 2 * H_);
  float* ctx_a = af32((size_t)B_ * A_);     _Float16* ctx_a16 = af16((size_t)B_ * A_);
  float* tcv = af32((size_t)B_ * H_);
  float* acv = af32((size_t)B_ * H_);
  float* cgb = af32(B_);
  float* gpb = af32(B_);
  float* ctx = af32((size_t)B_ * H_);
  _Float16* chcat16 = af16((size_t)B_ * 2 * H_);
  float* ch32 = af32((size_t)B_ * H_);
  _Float16* ch16 = af16((size_t)B_ * H_);
  float* logits = af32((size_t)B_ * V_);

  // ---- stage weights ----
  auto wcat = [&](int wih, int whh, int bih, int bhh, _Float16* wc, float* bc, int I, int Ip) {
    k_build_wcat<<<4096, 256, 0, stream>>>((const float*)d_in[wih], (const float*)d_in[whh],
                                           (const float*)d_in[bih], (const float*)d_in[bhh],
                                           wc, bc, I, Ip);
  };
  wcat(IP_REV_WIH_F, IP_REV_WHH_F, IP_REV_BIH_F, IP_REV_BHH_F, wc_rev_f, bc_rev_f, E_, EPAD);
  wcat(IP_REV_WIH_B, IP_REV_WHH_B, IP_REV_BIH_B, IP_REV_BHH_B, wc_rev_b, bc_rev_b, E_, EPAD);
  wcat(IP_SUM_WIH_F, IP_SUM_WHH_F, IP_SUM_BIH_F, IP_SUM_BHH_F, wc_sum_f, bc_sum_f, E_, EPAD);
  wcat(IP_SUM_WIH_B, IP_SUM_WHH_B, IP_SUM_BIH_B, IP_SUM_BHH_B, wc_sum_b, bc_sum_b, E_, EPAD);
  wcat(IP_DEC_WIH, IP_DEC_WHH, IP_DEC_BIH, IP_DEC_BHH, wc_dec, bc_dec, DI_, DIP);

  auto cast = [&](int idx, _Float16* dst, int rows, int cols) {
    k_cast_pad<<<nb((size_t)rows * cols), 256, 0, stream>>>((const float*)d_in[idx], dst, rows, cols, cols);
  };
  cast(IP_TF_W, w_tf, H_, 2 * H_);     cast(IP_MF_W, w_mf, 2 * H_, 4 * H_);
  cast(IP_EA_W, w_ea, H_, 2 * A_);     cast(IP_ATT_K, w_attk, H_, 2 * H_);
  cast(IP_ATT_Q, w_attq, H_, H_);      cast(IP_ATTA_K, w_attak, H_, A_);
  cast(IP_ATTA_Q, w_attaq, H_, H_);    cast(IP_TC_W, w_tc, H_, 2 * H_);
  cast(IP_AC_W, w_ac, H_, A_);         cast(IP_CHD_W, w_chd, H_, 2 * H_);
  cast(IP_GEN_W, w_gen, V_, H_);

  // ---- gather embeddings (time-major) ----
  k_gather_tm<<<nb((size_t)S_ * B_ * EPAD), 256, 0, stream>>>(P_emb, src, X_src, B_, B_, 0, S_);
  k_gather_tm<<<nb((size_t)LR_ * 160 * EPAD), 256, 0, stream>>>(P_emb, u_rev, X_rev, 160, 320, 0, LR_);
  k_gather_tm<<<nb((size_t)LR_ * 160 * EPAD), 256, 0, stream>>>(P_emb, p_rev, X_rev, 160, 320, 160, LR_);
  k_gather_tm<<<nb((size_t)LS_ * 160 * EPAD), 256, 0, stream>>>(P_emb, u_sum, X_sum, 160, 320, 0, LS_);
  k_gather_tm<<<nb((size_t)LS_ * 160 * EPAD), 256, 0, stream>>>(P_emb, p_sum, X_sum, 160, 320, 160, LS_);

  // ---- GRU scans (per-step fused WMMA GEMM + gate kernel) ----
  auto run_scan = [&](const _Float16* X, int Slen, int R,
                      const _Float16* wc, const float* bc,
                      _Float16* h16, float* h32, float* y, int dir) {
    hipMemsetAsync(h16, 0, (size_t)R * H_ * 2, stream);
    hipMemsetAsync(h32, 0, (size_t)R * H_ * 4, stream);
    for (int t = 0; t < Slen; t++) {
      int tx = dir ? (Slen - 1 - t) : t;
      gemm_launch(stream, X + (size_t)tx * R * EPAD, EPAD, EPAD, h16, H_,
                  wc, GRU_K, bc, preact, 2048, R, 2048, GRU_K, 0);
      int yoff = y ? (tx * 2 * H_ + dir * H_) : 0;
      k_gru_gate<<<nb((size_t)R * H_), 256, 0, stream>>>(preact, h16, h32, y,
                                                         S_ * 2 * H_, yoff, R);
    }
  };
  run_scan(X_src, S_, B_, wc_rev_f, bc_rev_f, h16_mf, h32_mf, enc_hid, 0);
  run_scan(X_src, S_, B_, wc_rev_b, bc_rev_b, h16_mb, h32_mb, enc_hid, 1);
  run_scan(X_rev, LR_, 320, wc_rev_f, bc_rev_f, h16_rf, h32_rf, nullptr, 0);
  run_scan(X_rev, LR_, 320, wc_rev_b, bc_rev_b, h16_rb, h32_rb, nullptr, 1);
  run_scan(X_sum, LS_, 320, wc_sum_f, bc_sum_f, h16_sf, h32_sf, nullptr, 0);
  run_scan(X_sum, LS_, 320, wc_sum_b, bc_sum_b, h16_sb, h32_sb, nullptr, 1);

  // ---- finals ----
  k_concat2_f32<<<nb((size_t)B_ * 2 * H_), 256, 0, stream>>>(h32_mf, H_, h32_mb, H_, enc_final, B_);
  k_concat2_f32<<<nb((size_t)320 * 2 * H_), 256, 0, stream>>>(h32_rf, H_, h32_rb, H_, rev_final, 320);
  k_concat2_f32<<<nb((size_t)320 * 2 * H_), 256, 0, stream>>>(h32_sf, H_, h32_sb, H_, sum_final, 320);

  // ---- memory network (MEM_LAYERS=1; queries are enc_final) ----
  k_mem_attn<<<B_, 256, 0, stream>>>(enc_final, rev_final, sum_final,
                                     (const float*)d_in[IP_RS1_W], (const float*)d_in[IP_RS1_B],
                                     (const float*)d_in[IP_RS2_W], (const float*)d_in[IP_RS2_B], u_mem);
  k_mem_attn<<<B_, 256, 0, stream>>>(enc_final, rev_final + (size_t)160 * 2 * H_,
                                     sum_final + (size_t)160 * 2 * H_,
                                     (const float*)d_in[IP_RS1_W], (const float*)d_in[IP_RS1_B],
                                     (const float*)d_in[IP_RS2_W], (const float*)d_in[IP_RS2_B], p_mem);
  k_concat2_f16<<<nb((size_t)B_ * 4 * H_), 256, 0, stream>>>(u_mem, 2 * H_, p_mem, 2 * H_, mem_cat16, B_);
  gemm_launch(stream, mem_cat16, 4 * H_, 4 * H_, nullptr, 0, w_mf, 4 * H_,
              (const float*)d_in[IP_MF_B], mem_out, 2 * H_, B_, 2 * H_, 4 * H_, 0);

  // ---- attrs / gates / h0 ----
  k_attrs<<<nb((size_t)B_ * 2 * A_), 256, 0, stream>>>((const float*)d_in[IP_USER_EMBED],
                                                       (const float*)d_in[IP_PRODUCT_EMBED],
                                                       user, prod, attrs);
  k_cast_pad<<<nb((size_t)B_ * 2 * A_), 256, 0, stream>>>(attrs, attrs16, B_, 2 * A_, 2 * A_);
  gemm_launch(stream, attrs16, 2 * A_, 2 * A_, nullptr, 0, w_ea, 2 * A_,
              (const float*)d_in[IP_EA_B], attr_final, H_, B_, H_, 2 * A_, 1);
  k_cast_pad<<<nb((size_t)B_ * 2 * H_), 256, 0, stream>>>(enc_final, enc_final16, B_, 2 * H_, 2 * H_);
  gemm_launch(stream, enc_final16, 2 * H_, 2 * H_, nullptr, 0, w_tf, 2 * H_,
              (const float*)d_in[IP_TF_B], text_final, H_, B_, H_, 2 * H_, 3);
  k_dotcat<<<B_, 256, 0, stream>>>((const float*)d_in[IP_EG_W], (const float*)d_in[IP_EG_B],
                                   enc_final, 2 * H_, attrs, 2 * A_, nullptr, 0, nullptr, 0, gbuf);
  k_h0<<<nb((size_t)B_ * H_), 256, 0, stream>>>(gbuf, text_final, attr_final, h_dec32, h_dec16);
  k_upm<<<nb((size_t)B_ * 1152), 256, 0, stream>>>(attrs, mem_out, upm);

  // ---- attention key projections ----
  k_cast_pad<<<nb((size_t)B_ * S_ * 2 * H_), 256, 0, stream>>>(enc_hid, enc_hid16, B_ * S_, 2 * H_, 2 * H_);
  gemm_launch(stream, enc_hid16, 2 * H_, 2 * H_, nullptr, 0, w_attk, 2 * H_, nullptr,
              pk, H_, B_ * S_, H_, 2 * H_, 0);
  // attrs16 [16x128] reinterpreted as [32x64] row-major
  gemm_launch(stream, attrs16, A_, A_, nullptr, 0, w_attak, A_, nullptr,
              pka, H_, 2 * B_, H_, A_, 0);

  hipMemsetAsync(ch32, 0, (size_t)B_ * H_ * 4, stream);

  // ---- decoder ----
  for (int t = 0; t < T_; t++) {
    k_dec_build<<<nb((size_t)B_ * DIP), 256, 0, stream>>>(P_emb, trg_, t, ch32, upm, decA016, prevemb);
    gemm_launch(stream, decA016, DIP, DIP, h_dec16, H_, wc_dec, DEC_K, bc_dec,
                preact, 2048, B_, 2048, DEC_K, 0);
    k_gru_gate<<<nb((size_t)B_ * H_), 256, 0, stream>>>(preact, h_dec16, h_dec32, nullptr, 0, 0, B_);
    gemm_launch(stream, h_dec16, H_, H_, nullptr, 0, w_attq, H_, nullptr, qp, H_, B_, H_, H_, 0);
    gemm_launch(stream, h_dec16, H_, H_, nullptr, 0, w_attaq, H_, nullptr, qpa, H_, B_, H_, H_, 0);
    k_attn_scores<<<dim3(B_, S_ / 8), 256, 0, stream>>>(qp, pk, (const float*)d_in[IP_ATT_E], src, sc);
    k_softmax_row<<<B_, 256, 0, stream>>>(sc, al, S_);
    k_attn_ctx<<<dim3(B_, (2 * H_) / 128), 128, 0, stream>>>(al, enc_hid, ctx_t);
    k_attr_attn<<<B_, 64, 0, stream>>>(qpa, pka, (const float*)d_in[IP_ATTA_E], attrs, ctx_a);
    k_cast_pad<<<nb((size_t)B_ * 2 * H_), 256, 0, stream>>>(ctx_t, ctx_t16, B_, 2 * H_, 2 * H_);
    gemm_launch(stream, ctx_t16, 2 * H_, 2 * H_, nullptr, 0, w_tc, 2 * H_,
                (const float*)d_in[IP_TC_B], tcv, H_, B_, H_, 2 * H_, 1);
    k_cast_pad<<<nb((size_t)B_ * A_), 256, 0, stream>>>(ctx_a, ctx_a16, B_, A_, A_);
    gemm_launch(stream, ctx_a16, A_, A_, nullptr, 0, w_ac, A_,
                (const float*)d_in[IP_AC_B], acv, H_, B_, H_, A_, 1);
    k_dotcat<<<B_, 256, 0, stream>>>((const float*)d_in[IP_CG_W], (const float*)d_in[IP_CG_B],
                                     ctx_t, 2 * H_, ctx_a, A_, h_dec32, H_, prevemb, E_, cgb);
    k_ctx_mix<<<nb((size_t)B_ * H_), 256, 0, stream>>>(cgb, tcv, acv, ctx);
    k_concat2_f16<<<nb((size_t)B_ * 2 * H_), 256, 0, stream>>>(h_dec32, H_, ctx, H_, chcat16, B_);
    gemm_launch(stream, chcat16, 2 * H_, 2 * H_, nullptr, 0, w_chd, 2 * H_,
                (const float*)d_in[IP_CHD_B], ch32, H_, B_, H_, 2 * H_, 1);
    k_cast_pad<<<nb((size_t)B_ * H_), 256, 0, stream>>>(ch32, ch16, B_, H_, H_);
    gemm_launch(stream, ch16, H_, H_, nullptr, 0, w_gen, H_, nullptr, logits, V_, B_, V_, H_, 0);
    k_dotcat<<<B_, 256, 0, stream>>>((const float*)d_in[IP_GP_W], (const float*)d_in[IP_GP_B],
                                     ctx, H_, h_dec32, H_, prevemb, E_, nullptr, 0, gpb);
    k_softmax_mix<<<B_, 256, 0, stream>>>(logits, gpb, out, t);
    k_copy_scatter<<<B_, 256, 0, stream>>>(al, gpb, src, out, t);
  }
}
